// CapsNet_83451214561483
// MI455X (gfx1250) — compile-verified, matrix operand order checked by reading, not run on hardware
//
#include <hip/hip_runtime.h>
#include <hip/hip_bf16.h>
#include <hip/hip_fp16.h>

typedef __attribute__((ext_vector_type(16))) _Float16 v16h;
typedef __attribute__((ext_vector_type(8)))  _Float16 v8h;
typedef __attribute__((ext_vector_type(8)))  float    v8f;
typedef _Float16 half_t;

#define BATCH 512
#define LDSS  72   // LDS row stride in halves: 144 B = 9*16 -> v8h loads stay 16B-aligned

// ---------------------------------------------------------------------------
// CDNA5 async global->LDS copy (ASYNCcnt path, CDNA5 ISA)
// ---------------------------------------------------------------------------
__device__ __forceinline__ void async_copy_b128(void* lds_dst, const void* gaddr) {
  unsigned lds = (unsigned)(uintptr_t)lds_dst;   // flat LDS addr: offset in [31:0]
  asm volatile("global_load_async_to_lds_b128 %0, %1, off"
               :: "v"(lds), "v"(gaddr) : "memory");
}
__device__ __forceinline__ void wait_async() {
  asm volatile("s_wait_asynccnt 0x0" ::: "memory");
}

// ---------------------------------------------------------------------------
// f32 -> f16 with zero padding: dst[Mp x Kp], src[M x K]
// ---------------------------------------------------------------------------
__global__ void pad_convert_kernel(const float* __restrict__ src,
                                   half_t* __restrict__ dst,
                                   int M, int K, int Kp, long total) {
  long i = (long)blockIdx.x * 256 + threadIdx.x;
  if (i >= total) return;
  long m = i / Kp; int k = (int)(i - m * Kp);
  dst[i] = (m < M && k < K) ? (half_t)src[m * (long)K + k] : (half_t)0.f;
}

__global__ void f32_to_f16_kernel(const float* __restrict__ src,
                                  half_t* __restrict__ dst, int n) {
  int i = blockIdx.x * 256 + threadIdx.x;
  if (i < n) dst[i] = (half_t)src[i];
}

// Fragment gather: 16 halves per lane as two aligned 16-byte LDS reads.
// ISA 16-bit A/B layout: lanes 0-15 kb=0, lanes 16-31 kb=8;
// halves 0..7 -> kb+0..7, halves 8..15 -> kb+16..23.
__device__ __forceinline__ v16h load_frag(const half_t* rowbase, int kb) {
  v8h lo = *(const v8h*)(rowbase + kb);
  v8h hi = *(const v8h*)(rowbase + kb + 16);
  v16h r;
  #pragma unroll
  for (int i = 0; i < 8; ++i) { r[i] = lo[i]; r[8 + i] = hi[i]; }
  return r;
}

#define WMMA_F16(A, B, C) \
  __builtin_amdgcn_wmma_f32_16x16x32_f16(false, (A), false, (B), (short)0, (C), false, false)

// ---------------------------------------------------------------------------
// Implicit-GEMM conv via WMMA.  M = OC(=256), N = B*OH*OW, K = IC*81 (Kp padded).
// Block: 256 thr = 8 waves; block tile 128(M) x 128(N); wave tile 32x64
// (2 A-frags x 4 B-frags -> 8 accumulators); K staged 64 -> 16 WMMA/barrier.
// A staged with async b128 DMA (overlaps the VALU im2col gather for B).
// ---------------------------------------------------------------------------
__launch_bounds__(256)
__global__ void conv_wmma_kernel(const half_t* __restrict__ W,   // [256, Kp] zero-padded
                                 const half_t* __restrict__ X,   // [B, IC, IH, IW]
                                 const float*  __restrict__ bias,// [256]
                                 half_t* __restrict__ out_f16,   // NCHW (or null)
                                 float*  __restrict__ out_f32,   // NCHW (or null)
                                 int IC, int IH, int IW,
                                 int OH, int OW, int stride, int relu,
                                 int K, int Kp) {
  const int PHW = OH * OW;

  __shared__ __align__(16) half_t As[128][LDSS];   // [m][k]  (128 x 64 used)
  __shared__ __align__(16) half_t Bs[128][LDSS];   // [n][k]  (transposed B)

  const int tid  = threadIdx.x;
  const int lane = tid & 31;
  const int wid  = tid >> 5;
  const int m0g  = blockIdx.y * 128;
  const int n0g  = blockIdx.x * 128;
  const int wm   = (wid >> 1) * 32;   // 4 wave rows  x 32
  const int wn   = (wid & 1)  * 64;   // 2 wave cols  x 64

  v8f acc[2][4] = {};
  const int kchunks = Kp >> 6;

  for (int kc = 0; kc < kchunks; ++kc) {
    const int k0 = kc << 6;

    // --- A stage: 128x64 halves as 1024 async b128 copies (4 per thread) ---
    #pragma unroll
    for (int j = 0; j < 4; ++j) {
      int e  = tid + j * 256;
      int ml = e >> 3, k8 = (e & 7) * 8;
      async_copy_b128(&As[ml][k8], &W[(long)(m0g + ml) * Kp + k0 + k8]);
    }
    // --- B stage (im2col gather), stored n-major: Bs[n][k] ---
    #pragma unroll 4
    for (int j = 0; j < 32; ++j) {
      int e = tid + j * 256;
      int nl = e & 127, kl = e >> 7;
      int k = k0 + kl;
      half_t v = (half_t)0.f;
      if (k < K) {
        int ic = k / 81; int r = k - ic * 81;
        int kh = r / 9;  int kw = r - kh * 9;
        int n  = n0g + nl;
        int b  = n / PHW; int p = n - b * PHW;
        int oh = p / OW;  int ow = p - oh * OW;
        int iy = oh * stride + kh, ix = ow * stride + kw;
        v = X[(((long)b * IC + ic) * IH + iy) * IW + ix];
      }
      Bs[nl][kl] = v;
    }
    wait_async();
    __syncthreads();

    const int row = wm + (lane & 15);
    const int col = wn + (lane & 15);
    const int kb  = (lane >> 4) << 3;
    #pragma unroll
    for (int kk = 0; kk < 64; kk += 32) {
      v16h a0 = load_frag(&As[row][kk],      kb);
      v16h a1 = load_frag(&As[row + 16][kk], kb);
      v16h b0 = load_frag(&Bs[col][kk],      kb);
      v16h b1 = load_frag(&Bs[col + 16][kk], kb);
      v16h b2 = load_frag(&Bs[col + 32][kk], kb);
      v16h b3 = load_frag(&Bs[col + 48][kk], kb);
      acc[0][0] = WMMA_F16(a0, b0, acc[0][0]);
      acc[0][1] = WMMA_F16(a0, b1, acc[0][1]);
      acc[0][2] = WMMA_F16(a0, b2, acc[0][2]);
      acc[0][3] = WMMA_F16(a0, b3, acc[0][3]);
      acc[1][0] = WMMA_F16(a1, b0, acc[1][0]);
      acc[1][1] = WMMA_F16(a1, b1, acc[1][1]);
      acc[1][2] = WMMA_F16(a1, b2, acc[1][2]);
      acc[1][3] = WMMA_F16(a1, b3, acc[1][3]);
    }
    __syncthreads();
  }

  // --- epilogue: C layout VGPR vv -> M = vv + 8*(lane>=16), N = lane&15 ---
  #pragma unroll
  for (int ni = 0; ni < 4; ++ni) {
    const int ncol = n0g + wn + ni * 16 + (lane & 15);
    const int b    = ncol / PHW;
    const int p    = ncol - b * PHW;
    #pragma unroll
    for (int mi = 0; mi < 2; ++mi) {
      #pragma unroll
      for (int vv = 0; vv < 8; ++vv) {
        int m = m0g + wm + mi * 16 + vv + ((lane >> 4) << 3);
        float cv = acc[mi][ni][vv] + bias[m];
        if (relu) cv = fmaxf(cv, 0.f);
        long idx = ((long)b * 256 + m) * PHW + p;   // NCHW, OC==256 both convs
        if (out_f16) out_f16[idx] = (half_t)cv;
        else         out_f32[idx] = cv;
      }
    }
  }
}

// ---------------------------------------------------------------------------
// Decoder GEMM via WMMA: C[M,512] = act(A[Mp,Kp] @ B[K,512] + bias)
// Same 128x128 block / 32x64 wave tiling.  N = 512 -> 4 column blocks.
// mode 0: ReLU, store f16 feature-major [M,512]
// mode 1: sigmoid, store f32 transposed out[n*784 + m]
// ---------------------------------------------------------------------------
__launch_bounds__(256)
__global__ void gemm_wmma_kernel(const half_t* __restrict__ A,
                                 const half_t* __restrict__ Bm,
                                 const float*  __restrict__ bias,
                                 half_t* __restrict__ out_f16,
                                 float*  __restrict__ out_sig,
                                 int M, int K, int Kp, int mode) {
  __shared__ __align__(16) half_t As[128][LDSS];
  __shared__ __align__(16) half_t Bs[128][LDSS];

  const int tid  = threadIdx.x;
  const int lane = tid & 31;
  const int wid  = tid >> 5;
  const int m0g  = blockIdx.y * 128;
  const int n0g  = blockIdx.x * 128;
  const int wm   = (wid >> 1) * 32;
  const int wn   = (wid & 1)  * 64;

  v8f acc[2][4] = {};
  const int kchunks = Kp >> 6;

  for (int kc = 0; kc < kchunks; ++kc) {
    const int k0 = kc << 6;
    #pragma unroll
    for (int j = 0; j < 4; ++j) {
      int e  = tid + j * 256;
      int ml = e >> 3, k8 = (e & 7) * 8;
      async_copy_b128(&As[ml][k8], &A[(long)(m0g + ml) * Kp + k0 + k8]);
    }
    #pragma unroll
    for (int j = 0; j < 32; ++j) {
      int e = tid + j * 256;
      int nl = e & 127, kl = e >> 7;
      int k = k0 + kl;
      Bs[nl][kl] = (k < K) ? Bm[(long)k * BATCH + n0g + nl] : (half_t)0.f;
    }
    wait_async();
    __syncthreads();

    const int row = wm + (lane & 15);
    const int col = wn + (lane & 15);
    const int kb  = (lane >> 4) << 3;
    #pragma unroll
    for (int kk = 0; kk < 64; kk += 32) {
      v16h a0 = load_frag(&As[row][kk],      kb);
      v16h a1 = load_frag(&As[row + 16][kk], kb);
      v16h b0 = load_frag(&Bs[col][kk],      kb);
      v16h b1 = load_frag(&Bs[col + 16][kk], kb);
      v16h b2 = load_frag(&Bs[col + 32][kk], kb);
      v16h b3 = load_frag(&Bs[col + 48][kk], kb);
      acc[0][0] = WMMA_F16(a0, b0, acc[0][0]);
      acc[0][1] = WMMA_F16(a0, b1, acc[0][1]);
      acc[0][2] = WMMA_F16(a0, b2, acc[0][2]);
      acc[0][3] = WMMA_F16(a0, b3, acc[0][3]);
      acc[1][0] = WMMA_F16(a1, b0, acc[1][0]);
      acc[1][1] = WMMA_F16(a1, b1, acc[1][1]);
      acc[1][2] = WMMA_F16(a1, b2, acc[1][2]);
      acc[1][3] = WMMA_F16(a1, b3, acc[1][3]);
    }
    __syncthreads();
  }

  #pragma unroll
  for (int ni = 0; ni < 4; ++ni) {
    const int ncol = n0g + wn + ni * 16 + (lane & 15);
    #pragma unroll
    for (int mi = 0; mi < 2; ++mi) {
      #pragma unroll
      for (int vv = 0; vv < 8; ++vv) {
        int m = m0g + wm + mi * 16 + vv + ((lane >> 4) << 3);
        if (m < M) {
          float cv = acc[mi][ni][vv] + bias[m];
          if (mode == 0) {
            cv = fmaxf(cv, 0.f);
            out_f16[(long)m * BATCH + ncol] = (half_t)cv;
          } else {
            out_sig[(long)ncol * 784 + m] = 1.f / (1.f + expf(-cv));
          }
        }
      }
    }
  }
}

// ---------------------------------------------------------------------------
// squash over 8-dim primary capsules: p [B,9216] -> caps [B,1152,8]
// ---------------------------------------------------------------------------
__global__ void squash_caps_kernel(const float* __restrict__ p,
                                   float* __restrict__ caps) {
  long t = (long)blockIdx.x * 256 + threadIdx.x;   // b*1152 + n
  if (t >= (long)BATCH * 1152) return;
  float x[8]; float sq = 0.f;
  #pragma unroll
  for (int i = 0; i < 8; ++i) { x[i] = p[t * 8 + i]; sq += x[i] * x[i]; }
  float scale = (sq / (1.f + sq)) / (sqrtf(sq) + 1e-8f);
  #pragma unroll
  for (int i = 0; i < 8; ++i) caps[t * 8 + i] = x[i] * scale;
}

// ---------------------------------------------------------------------------
// u_hat[b,o,n,d] = sum_i W_route[o,n,d,i] * caps[b,n,i]   (K=8: VALU,
// bandwidth-bound on the 377 MB u_hat stream; matrix pipe would be 75% idle)
// ---------------------------------------------------------------------------
__launch_bounds__(256)
__global__ void uhat_kernel(const float* __restrict__ Wr,
                            const float* __restrict__ caps,
                            float* __restrict__ uhat) {
  const int on = blockIdx.x;          // o*1152 + n
  const int o  = on / 1152;
  const int n  = on - o * 1152;
  __shared__ float Ws[128];           // [d][i]
  if (threadIdx.x < 128) Ws[threadIdx.x] = Wr[(long)on * 128 + threadIdx.x];
  __syncthreads();
  for (int b = threadIdx.x; b < BATCH; b += 256) {
    float ci[8];
    #pragma unroll
    for (int i = 0; i < 8; ++i) ci[i] = caps[((long)b * 1152 + n) * 8 + i];
    float* up = uhat + (((long)b * 10 + o) * 1152 + n) * 16;
    #pragma unroll
    for (int d = 0; d < 16; ++d) {
      float s = 0.f;
      #pragma unroll
      for (int i = 0; i < 8; ++i) s += Ws[d * 8 + i] * ci[i];
      up[d] = s;
    }
  }
}

// ---------------------------------------------------------------------------
// Routing step: c = softmax_o(b_logit); s[b,o,:] = sum_n c*u_hat; v = squash(s)
// ---------------------------------------------------------------------------
__launch_bounds__(256)
__global__ void routing_cs_kernel(const float* __restrict__ bl,
                                  const float* __restrict__ uhat,
                                  float* __restrict__ vout) {
  const int bo = blockIdx.x;          // b*10 + o
  const int b  = bo / 10;
  const int o  = bo - b * 10;
  __shared__ float red[16 * 256];
  const int tid = threadIdx.x;

  float acc[16];
  #pragma unroll
  for (int d = 0; d < 16; ++d) acc[d] = 0.f;

  for (int n = tid; n < 1152; n += 256) {
    const float* blp = bl + (long)b * 10 * 1152 + n;
    float vals[10]; float mx = -1e30f;
    #pragma unroll
    for (int oo = 0; oo < 10; ++oo) {
      vals[oo] = blp[oo * 1152];
      mx = fmaxf(mx, vals[oo]);
    }
    float den = 0.f;
    #pragma unroll
    for (int oo = 0; oo < 10; ++oo) den += expf(vals[oo] - mx);
    float c = expf(vals[o] - mx) / den;
    const float* up = uhat + ((long)bo * 1152 + n) * 16;
    #pragma unroll
    for (int d = 0; d < 16; ++d) acc[d] += c * up[d];
  }
  #pragma unroll
  for (int d = 0; d < 16; ++d) red[d * 256 + tid] = acc[d];
  __syncthreads();
  for (int s = 128; s > 0; s >>= 1) {
    if (tid < s) {
      #pragma unroll
      for (int d = 0; d < 16; ++d) red[d * 256 + tid] += red[d * 256 + tid + s];
    }
    __syncthreads();
  }
  if (tid == 0) {
    float sq = 0.f;
    #pragma unroll
    for (int d = 0; d < 16; ++d) { float sd = red[d * 256]; sq += sd * sd; }
    float scale = (sq / (1.f + sq)) / (sqrtf(sq) + 1e-8f);
    #pragma unroll
    for (int d = 0; d < 16; ++d) vout[(long)bo * 16 + d] = red[d * 256] * scale;
  }
}

// b_logit[b,o,n] += sum_d u_hat[b,o,n,d] * v[b,o,d]
__global__ void routing_update_kernel(const float* __restrict__ uhat,
                                      const float* __restrict__ v,
                                      float* __restrict__ bl) {
  long t = (long)blockIdx.x * 256 + threadIdx.x;   // (b*10+o)*1152 + n
  if (t >= (long)BATCH * 10 * 1152) return;
  long bo = t / 1152;
  const float* up = uhat + t * 16;
  const float* vp = v + bo * 16;
  float s = 0.f;
  #pragma unroll
  for (int d = 0; d < 16; ++d) s += up[d] * vp[d];
  bl[t] += s;
}

// logits = ||v||; argmax mask; dec_in [160, 512] f16 feature-major
__launch_bounds__(192)
__global__ void head_kernel(const float* __restrict__ v,
                            float* __restrict__ logits,
                            half_t* __restrict__ dec_in) {
  const int b = blockIdx.x;
  __shared__ float nrm[10];
  __shared__ int amax;
  const int t = threadIdx.x;
  if (t < 10) {
    float sq = 0.f;
    #pragma unroll
    for (int d = 0; d < 16; ++d) {
      float x = v[((long)b * 10 + t) * 16 + d];
      sq += x * x;
    }
    float l = sqrtf(sq);
    nrm[t] = l;
    logits[b * 10 + t] = l;
  }
  __syncthreads();
  if (t == 0) {
    int am = 0; float bst = nrm[0];
    for (int o = 1; o < 10; ++o) if (nrm[o] > bst) { bst = nrm[o]; am = o; }
    amax = am;
  }
  __syncthreads();
  if (t < 160) {
    int o = t >> 4, d = t & 15;
    float x = (o == amax) ? v[((long)b * 10 + o) * 16 + d] : 0.f;
    dec_in[(long)t * BATCH + b] = (half_t)x;
  }
}

// ---------------------------------------------------------------------------
extern "C" void kernel_launch(void* const* d_in, const int* in_sizes, int n_in,
                              void* d_out, int out_size, void* d_ws, size_t ws_size,
                              hipStream_t stream) {
  const float* x      = (const float*)d_in[0];
  const float* conv_w = (const float*)d_in[1];
  const float* conv_b = (const float*)d_in[2];
  const float* pc_w   = (const float*)d_in[3];
  const float* pc_b   = (const float*)d_in[4];
  const float* Wroute = (const float*)d_in[5];
  const float* dw1    = (const float*)d_in[6];
  const float* db1    = (const float*)d_in[7];
  const float* dw2    = (const float*)d_in[8];
  const float* db2    = (const float*)d_in[9];
  const float* dw3    = (const float*)d_in[10];
  const float* db3    = (const float*)d_in[11];
  float* out = (float*)d_out;

  // workspace cursor
  char* ws = (char*)d_ws;
  auto alloc = [&](size_t bytes) -> void* {
    void* p = (void*)ws;
    ws += (bytes + 255) & ~(size_t)255;
    return p;
  };
  half_t* x_h    = (half_t*)alloc(401408u * 2);            // [512,1,28,28] f16
  half_t* cw_h   = (half_t*)alloc(32768u  * 2);            // conv1 w [256,128] padded
  half_t* pcw_h  = (half_t*)alloc(5308416u * 2);           // pc w [256,20736]
  half_t* w1_h   = (half_t*)alloc(98304u  * 2);            // [512,192] padded
  half_t* w2_h   = (half_t*)alloc(524288u * 2);            // [1024,512]
  half_t* w3_h   = (half_t*)alloc(917504u * 2);            // [896,1024] padded rows
  half_t* h_h    = (half_t*)alloc(52428800ull * 2);        // conv1 out f16 NCHW
  float*  p_f    = (float*)alloc(4718592ull * 4);          // pc out f32 NCHW
  float*  caps_f = (float*)alloc(4718592ull * 4);          // squashed caps
  float*  uhat_f = (float*)alloc(94371840ull * 4);         // [B,10,1152,16]
  float*  bl_f   = (float*)alloc(5898240ull * 4);          // b_logit
  float*  v_f    = (float*)alloc(81920u * 4);              // [B,10,16]
  half_t* din_h  = (half_t*)alloc(81920u * 2);             // dec_in [160,512]
  half_t* h1_h   = (half_t*)alloc(262144u * 2);            // [512,512]
  half_t* h2_h   = (half_t*)alloc(524288u * 2);            // [1024,512]

  auto pad = [&](const float* s, half_t* d, int M, int K, int Kp, long total) {
    pad_convert_kernel<<<(int)((total + 255) / 256), 256, 0, stream>>>(s, d, M, K, Kp, total);
  };
  f32_to_f16_kernel<<<(401408 + 255) / 256, 256, 0, stream>>>(x, x_h, 401408);
  pad(conv_w, cw_h, 256, 81, 128, 32768);
  pad(pc_w, pcw_h, 256, 20736, 20736, 5308416);
  pad(dw1, w1_h, 512, 160, 192, 98304);
  pad(dw2, w2_h, 1024, 512, 512, 524288);
  pad(dw3, w3_h, 784, 1024, 1024, 917504);

  // conv1: [512,1,28,28] -> relu -> [512,256,20,20] f16.  N=204800/128=1600
  conv_wmma_kernel<<<dim3(1600, 2), 256, 0, stream>>>(
      cw_h, x_h, conv_b, h_h, nullptr, 1, 28, 28, 20, 20, 1, 1, 81, 128);

  // primary caps: [512,256,20,20] -> [512,256,6,6] f32.  N=18432/128=144
  conv_wmma_kernel<<<dim3(144, 2), 256, 0, stream>>>(
      pcw_h, h_h, pc_b, nullptr, p_f, 256, 20, 20, 6, 6, 2, 0, 20736, 20736);

  // squash primary capsules
  squash_caps_kernel<<<(BATCH * 1152 + 255) / 256, 256, 0, stream>>>(p_f, caps_f);

  // u_hat
  uhat_kernel<<<10 * 1152, 256, 0, stream>>>(Wroute, caps_f, uhat_f);

  // dynamic routing
  hipMemsetAsync(bl_f, 0, 5898240ull * 4, stream);
  const int upd_blocks = (int)(((long)BATCH * 10 * 1152 + 255) / 256);
  for (int it = 0; it < 2; ++it) {
    routing_cs_kernel<<<BATCH * 10, 256, 0, stream>>>(bl_f, uhat_f, v_f);
    routing_update_kernel<<<upd_blocks, 256, 0, stream>>>(uhat_f, v_f, bl_f);
  }
  routing_cs_kernel<<<BATCH * 10, 256, 0, stream>>>(bl_f, uhat_f, v_f);

  // logits + argmax mask -> dec_in
  head_kernel<<<BATCH, 192, 0, stream>>>(v_f, out, din_h);

  // decoder MLP (WMMA GEMMs, N = batch = 512 -> 4 column blocks)
  gemm_wmma_kernel<<<dim3(4, 4), 256, 0, stream>>>(    // 512 x 160
      w1_h, din_h, db1, h1_h, nullptr, 512, 160, 192, 0);
  gemm_wmma_kernel<<<dim3(4, 8), 256, 0, stream>>>(    // 1024 x 512
      w2_h, h1_h, db2, h2_h, nullptr, 1024, 512, 512, 0);
  gemm_wmma_kernel<<<dim3(4, 7), 256, 0, stream>>>(    // 784 x 1024, sigmoid
      w3_h, h2_h, db3, nullptr, out + BATCH * 10, 784, 1024, 1024, 1);
}
